// ExtAttention_73426760892659
// MI455X (gfx1250) — matrix operand order repaired and checked by reading, compiled, not run on hardware
//
#include <hip/hip_runtime.h>
#include <hip/hip_bf16.h>
#include <stdint.h>

typedef __attribute__((ext_vector_type(16))) __bf16 bf16x16;
typedef __attribute__((ext_vector_type(8)))  __bf16 bf16x8;
typedef __attribute__((ext_vector_type(8)))  float  f32x8;

#define BATCH 4
#define DIM 256
#define NSEQ 2048
#define HEADS 4
#define DHEAD 32
#define HID 128
#define INDC 5
#define NB 64       // j-block width in attention
#define BPAD 68     // padded bias row stride (floats): 8-row offset = 32-bank shift -> conflict-free

// A-fragment (16x32, 16-bit): lane<16 -> row=lane, K {0..15}; lane>=16 -> row=lane-16, K {16..31}
__device__ __forceinline__ bf16x16 load_afrag(const __bf16* base, int rowStride, int lane) {
    int kh  = lane >> 4;
    int row = lane & 15;
    const __bf16* p = base + (size_t)row * rowStride + kh * 16;
    bf16x16 a;
    *(bf16x8*)&a       = *(const bf16x8*)p;
    *((bf16x8*)&a + 1) = *(const bf16x8*)(p + 8);
    return a;
}

// B-fragment (32x16, 16-bit): lane holds its column's contiguous 16-element K-run (caller computes p)
__device__ __forceinline__ bf16x16 load_bfrag16(const __bf16* p) {
    bf16x16 v;
    *(bf16x8*)&v       = *(const bf16x8*)p;
    *((bf16x8*)&v + 1) = *(const bf16x8*)(p + 8);
    return v;
}

__global__ void cvt_bf16(const float* __restrict__ in, __bf16* __restrict__ out, int n) {
    int i = blockIdx.x * blockDim.x + threadIdx.x;
    if (i < n) out[i] = (__bf16)in[i];
}

// x (b, c, n) f32 -> xT (b, n, c) bf16, 32x32 LDS tiles
__global__ void transpose_x(const float* __restrict__ x, __bf16* __restrict__ xT) {
    __shared__ float tile[32][33];
    int b = blockIdx.z, i0 = blockIdx.x * 32, c0 = blockIdx.y * 32;
    int tx = threadIdx.x, ty = threadIdx.y; // 32 x 8
    const float* xp = x + ((size_t)b * DIM + c0) * NSEQ + i0;
    for (int r = ty; r < 32; r += 8)
        tile[r][tx] = xp[(size_t)r * NSEQ + tx];
    __syncthreads();
    __bf16* op = xT + ((size_t)b * NSEQ + i0) * DIM + c0;
    for (int r = ty; r < 32; r += 8)
        op[(size_t)r * DIM + tx] = (__bf16)tile[tx][r];
}

// qkv = w_qkv @ x : per wave one 16(o) x 16(i) tile, K=256 in 8 WMMA steps.
__global__ __launch_bounds__(256) void qkv_gemm(
    const __bf16* __restrict__ wq, const __bf16* __restrict__ xT,
    __bf16* __restrict__ qb, __bf16* __restrict__ kb, __bf16* __restrict__ vb)
{
    const int lane = threadIdx.x & 31, wave = threadIdx.x >> 5;
    const int i0 = (blockIdx.x * 8 + wave) * 16;
    const int o0 = blockIdx.y * 16;
    const int b  = blockIdx.z;
    const int hi = lane >> 4, colL = lane & 15;
    f32x8 c = {};
    const __bf16* xrow = xT + ((size_t)b * NSEQ + i0 + colL) * DIM + hi * 16;
#pragma unroll
    for (int cb = 0; cb < DIM / 32; ++cb) {
        bf16x16 a  = load_afrag(wq + (size_t)o0 * DIM + cb * 32, DIM, lane);
        bf16x16 bx = load_bfrag16(xrow + cb * 32);
        c = __builtin_amdgcn_wmma_f32_16x16x32_bf16(false, a, false, bx, (short)0, c, false, false);
    }
    const float qscale = 0.17677669529663687f; // 32^-0.5
#pragma unroll
    for (int v = 0; v < 8; ++v) {
        int og = o0 + v + hi * 8;
        int ig = i0 + colL;
        float val = c[v];
        if (og < 128) {
            int hh = og >> 5, d = og & 31;
            qb[(((size_t)(b * HEADS + hh)) * NSEQ + ig) * DHEAD + d] = (__bf16)(val * qscale);
        } else if (og < 256) {
            int o2 = og - 128, hh = o2 >> 5, d = o2 & 31;
            kb[(((size_t)(b * HEADS + hh)) * NSEQ + ig) * DHEAD + d] = (__bf16)val;
        } else {
            int o2 = og - 256, hh = o2 >> 5, d = o2 & 31;
            vb[(((size_t)(b * HEADS + hh)) * DHEAD + d) * NSEQ + ig] = (__bf16)val;
        }
    }
}

// Fused attention: one workgroup = (b, 32 query rows) x ALL 4 heads (8 waves: wave=(h,strip)).
// Per j-block: async-DMA the raw indicator tile to LDS (ASYNCcnt), cooperatively fold the
// 5 channels into per-head biases (read once -> 4 heads), then WMMA S, online softmax, WMMA PV.
__global__ __launch_bounds__(256) void attn_kernel(
    const __bf16* __restrict__ qb, const __bf16* __restrict__ kb,
    const __bf16* __restrict__ vb, const float* __restrict__ ind,
    const float* __restrict__ wind, __bf16* __restrict__ aoT)
{
    __shared__ float  rawind[INDC][32][64];         // 40KB raw indicator tile (per-thread private slots)
    __shared__ float  biasb[2][HEADS][32][BPAD];    // 68KB double-buffered per-head bias
    __shared__ __bf16 pbuf[8][16][NB];              // 16KB per-wave P staging (C-layout -> A-layout)

    const int tid   = threadIdx.x;
    const int lane  = tid & 31;
    const int wave  = tid >> 5;
    const int h     = wave >> 1;
    const int strip = wave & 1;
    const int b     = blockIdx.y;
    const int iBase = blockIdx.x * 32;
    const int i0    = iBase + strip * 16;
    const int hi    = lane >> 4, col = lane & 15;

    // cooperative bias-phase mapping: thread -> (rows r0, r0+16) x j-quad jq
    const int jq = tid & 15;
    const int r0 = tid >> 4;

    const __bf16* qrow = qb + (((size_t)(b * HEADS + h)) * NSEQ + i0) * DHEAD;
    bf16x16 aq = load_afrag(qrow, DHEAD, lane);

    float wiall[HEADS][INDC];
#pragma unroll
    for (int hh = 0; hh < HEADS; ++hh)
#pragma unroll
        for (int cc = 0; cc < INDC; ++cc) wiall[hh][cc] = wind[hh * INDC + cc];

    const size_t cstride = (size_t)NSEQ * NSEQ;
    const float* gIndBase = ind + (size_t)b * INDC * cstride + (size_t)iBase * NSEQ;
    const __bf16* kbase = kb + ((size_t)(b * HEADS + h)) * NSEQ * DHEAD;
    const __bf16* vbase = vb + ((size_t)(b * HEADS + h)) * DHEAD * NSEQ;

    const unsigned rawBase = (unsigned)(uintptr_t)&rawind[0][0][0];

    f32x8 oacc[2] = {f32x8{}, f32x8{}};
    float mrow[8], lrow[8];
#pragma unroll
    for (int v = 0; v < 8; ++v) { mrow[v] = -INFINITY; lrow[v] = 0.f; }

    for (int jb = 0; jb < NSEQ; jb += NB) {
        const int pb = (jb >> 6) & 1;

        // ---- phase A: async copy this thread's raw-indicator chunks (global -> LDS, ASYNCcnt) ----
        {
            const uint64_t gbase = (uint64_t)(uintptr_t)(gIndBase + jb);
#pragma unroll
            for (int rr = 0; rr < 2; ++rr) {
                const int r = r0 + rr * 16;
#pragma unroll
                for (int cc = 0; cc < INDC; ++cc) {
                    unsigned voff = (unsigned)((cc * cstride + (size_t)r * NSEQ + jq * 4) * sizeof(float));
                    unsigned loff = rawBase + (unsigned)(((cc * 32 + r) * 64 + jq * 4) * sizeof(float));
                    asm volatile("global_load_async_to_lds_b128 %0, %1, %2"
                                 :: "v"(loff), "v"(voff), "s"(gbase) : "memory");
                }
            }
            asm volatile("s_wait_asynccnt 0" ::: "memory");
        }

        // ---- phase B: fold 5 channels into 4 per-head biases (read raw once per (i,j)) ----
#pragma unroll
        for (int rr = 0; rr < 2; ++rr) {
            const int r = r0 + rr * 16;
            float4 ch[INDC];
#pragma unroll
            for (int cc = 0; cc < INDC; ++cc)
                ch[cc] = *(const float4*)&rawind[cc][r][jq * 4];
#pragma unroll
            for (int hh = 0; hh < HEADS; ++hh) {
                float4 bs;
                bs.x = bs.y = bs.z = bs.w = 0.f;
#pragma unroll
                for (int cc = 0; cc < INDC; ++cc) {
                    const float w = wiall[hh][cc];
                    bs.x += w * ch[cc].x; bs.y += w * ch[cc].y;
                    bs.z += w * ch[cc].z; bs.w += w * ch[cc].w;
                }
                *(float4*)&biasb[pb][hh][r][jq * 4] = bs;
            }
        }
        __syncthreads();

        // ---- phase C: S = q k^T + bias, online softmax, O += P v^T ----
        f32x8 s[4];
#pragma unroll
        for (int t = 0; t < 4; ++t) {
            const int j = jb + t * 16 + col;
            bf16x16 bk = load_bfrag16(kbase + (size_t)j * DHEAD + hi * 16);
            f32x8 zero = {};
            s[t] = __builtin_amdgcn_wmma_f32_16x16x32_bf16(false, aq, false, bk, (short)0, zero, false, false);
#pragma unroll
            for (int v = 0; v < 8; ++v)
                s[t][v] += biasb[pb][h][strip * 16 + v + hi * 8][t * 16 + col];
        }
#pragma unroll
        for (int v = 0; v < 8; ++v) {
            float m = s[0][v];
#pragma unroll
            for (int t = 1; t < 4; ++t) m = fmaxf(m, s[t][v]);
#pragma unroll
            for (int off = 1; off < 16; off <<= 1) m = fmaxf(m, __shfl_xor(m, off, 32));
            const float mnew  = fmaxf(mrow[v], m);
            const float alpha = __expf(mrow[v] - mnew);
            mrow[v] = mnew;
            float ls = 0.f;
#pragma unroll
            for (int t = 0; t < 4; ++t) {
                const float p = __expf(s[t][v] - mnew);
                s[t][v] = p;
                ls += p;
            }
#pragma unroll
            for (int off = 1; off < 16; off <<= 1) ls += __shfl_xor(ls, off, 32);
            lrow[v] = lrow[v] * alpha + ls;
            oacc[0][v] *= alpha;
            oacc[1][v] *= alpha;
        }
#pragma unroll
        for (int t = 0; t < 4; ++t)
#pragma unroll
            for (int v = 0; v < 8; ++v)
                pbuf[wave][v + hi * 8][t * 16 + col] = (__bf16)s[t][v];
#pragma unroll
        for (int chk = 0; chk < 2; ++chk) {
            bf16x16 ap = load_afrag(&pbuf[wave][0][chk * 32], NB, lane);
#pragma unroll
            for (int dh2 = 0; dh2 < 2; ++dh2) {
                const __bf16* vp = vbase + (size_t)(dh2 * 16 + col) * NSEQ + jb + chk * 32 + hi * 16;
                bf16x16 bv = load_bfrag16(vp);
                oacc[dh2] = __builtin_amdgcn_wmma_f32_16x16x32_bf16(false, ap, false, bv, (short)0, oacc[dh2], false, false);
            }
        }
    }

    // normalize + store attention output transposed: aoT[b][i][h*32+d]
#pragma unroll
    for (int dh2 = 0; dh2 < 2; ++dh2)
#pragma unroll
        for (int v = 0; v < 8; ++v) {
            const float val = oacc[dh2][v] / lrow[v];
            const int i = i0 + v + hi * 8;
            const int d = dh2 * 16 + col;
            aoT[((size_t)b * NSEQ + i) * HID + h * DHEAD + d] = (__bf16)val;
        }
}

// out = w_out @ attn_out + b_out
__global__ __launch_bounds__(256) void out_gemm(
    const __bf16* __restrict__ wo, const __bf16* __restrict__ aoT,
    const float* __restrict__ bout, float* __restrict__ out)
{
    const int lane = threadIdx.x & 31, wave = threadIdx.x >> 5;
    const int i0 = (blockIdx.x * 8 + wave) * 16;
    const int o0 = blockIdx.y * 16;
    const int b  = blockIdx.z;
    const int hi = lane >> 4, colL = lane & 15;
    f32x8 c = {};
    const __bf16* arow = aoT + ((size_t)b * NSEQ + i0 + colL) * HID + hi * 16;
#pragma unroll
    for (int cb = 0; cb < HID / 32; ++cb) {
        bf16x16 a  = load_afrag(wo + (size_t)o0 * HID + cb * 32, HID, lane);
        bf16x16 bx = load_bfrag16(arow + cb * 32);
        c = __builtin_amdgcn_wmma_f32_16x16x32_bf16(false, a, false, bx, (short)0, c, false, false);
    }
#pragma unroll
    for (int v = 0; v < 8; ++v) {
        int og = o0 + v + hi * 8;
        int ig = i0 + colL;
        out[((size_t)b * DIM + og) * NSEQ + ig] = c[v] + bout[og];
    }
}

extern "C" void kernel_launch(void* const* d_in, const int* in_sizes, int n_in,
                              void* d_out, int out_size, void* d_ws, size_t ws_size,
                              hipStream_t stream) {
    const float* x    = (const float*)d_in[0];
    const float* ind  = (const float*)d_in[1];
    const float* wqkv = (const float*)d_in[2];
    const float* wind = (const float*)d_in[3];
    const float* wout = (const float*)d_in[4];
    const float* bout = (const float*)d_in[5];

    char* ws = (char*)d_ws;
    __bf16* xT  = (__bf16*)(ws + 0);              // 4 MB
    __bf16* wqb = (__bf16*)(ws + 4194304);        // 192 KB
    __bf16* wob = (__bf16*)(ws + 4390912);        // 64 KB
    __bf16* qb  = (__bf16*)(ws + 4456448);        // 2 MB
    __bf16* kb  = (__bf16*)(ws + 6553600);        // 2 MB
    __bf16* vb  = (__bf16*)(ws + 8650752);        // 2 MB
    __bf16* aoT = (__bf16*)(ws + 10747904);       // 2 MB

    cvt_bf16<<<(3 * HID * DIM + 255) / 256, 256, 0, stream>>>(wqkv, wqb, 3 * HID * DIM);
    cvt_bf16<<<(DIM * HID + 255) / 256, 256, 0, stream>>>(wout, wob, DIM * HID);
    transpose_x<<<dim3(NSEQ / 32, DIM / 32, BATCH), dim3(32, 8), 0, stream>>>(x, xT);
    qkv_gemm<<<dim3(NSEQ / 128, (3 * HID) / 16, BATCH), 256, 0, stream>>>(wqb, xT, qb, kb, vb);
    attn_kernel<<<dim3(NSEQ / 32, BATCH), 256, 0, stream>>>(qb, kb, vb, ind, wind, aoT);
    out_gemm<<<dim3(NSEQ / 128, DIM / 16, BATCH), 256, 0, stream>>>(wob, aoT, bout, (float*)d_out);
}
